// TransformerBlock_4595615006979
// MI455X (gfx1250) — matrix operand order repaired and checked by reading, not compiled, not run on hardware
//
#include <hip/hip_runtime.h>
#include <hip/hip_bf16.h>
#include <cstdint>

// ---------------------------------------------------------------------------
// Types
// ---------------------------------------------------------------------------
typedef __attribute__((ext_vector_type(16))) __bf16   v16bf;
typedef __attribute__((ext_vector_type(8)))  float    v8f;
typedef __attribute__((ext_vector_type(4)))  unsigned u32x4;
typedef unsigned short ushort;

union FragU { u32x4 u[2]; v16bf v; };

static constexpr int S_DIM = 4096;
static constexpr int D_DIM = 1024;
static constexpr int F_DIM = 4096;

__device__ __forceinline__ ushort f32_to_bf16(float f) {
    unsigned u = __float_as_uint(f);
    unsigned r = u + 0x7FFFu + ((u >> 16) & 1u);   // round-to-nearest-even
    return (ushort)(r >> 16);
}
__device__ __forceinline__ float bf16_to_f32(ushort h) {
    return __uint_as_float(((unsigned)h) << 16);
}
__device__ __forceinline__ ushort to_bf16_elem(float v)  { return f32_to_bf16(v); }
__device__ __forceinline__ ushort to_bf16_elem(ushort v) { return v; }

// Async global->LDS 16B copy (CDNA5 VFLAT async, tracked by ASYNCcnt).
__device__ __forceinline__ void async_copy_b128(unsigned lds_off, const void* gaddr) {
    asm volatile("global_load_async_to_lds_b128 %0, %1, off"
                 :: "v"(lds_off), "v"((unsigned long long)(uintptr_t)gaddr)
                 : "memory");
}
__device__ __forceinline__ void wait_async_le4() {
    asm volatile("s_wait_asynccnt 0x4" ::: "memory");
}
__device__ __forceinline__ void wait_async_le0() {
    asm volatile("s_wait_asynccnt 0x0" ::: "memory");
}
__device__ __forceinline__ unsigned lds_addr_of(const void* p) {
    // Generic LDS addresses carry the LDS byte offset in the low 32 bits
    // (flat aperture truncation, ISA 10.2).
    return (unsigned)(uintptr_t)p;
}

// ---------------------------------------------------------------------------
// f32 -> bf16 conversion (grid-stride)
// ---------------------------------------------------------------------------
__global__ __launch_bounds__(256)
void cvt_f32_to_bf16(const float* __restrict__ src, ushort* __restrict__ dst, int n) {
    for (int i = blockIdx.x * blockDim.x + threadIdx.x; i < n; i += gridDim.x * blockDim.x)
        dst[i] = f32_to_bf16(src[i]);
}

// ---------------------------------------------------------------------------
// Fused transpose(+convert) to bf16:  out[c][r] = bf16(in[r][c])
// 32x32 LDS tile, coalesced on both sides. T = float (convert) or ushort.
// ---------------------------------------------------------------------------
template <typename T>
__global__ __launch_bounds__(256)
void transpose_to_bf16(const T* __restrict__ in, ushort* __restrict__ out, int R, int C)
{
    __shared__ ushort tile[32][33];
    const int tx = threadIdx.x & 31;
    const int ty = threadIdx.x >> 5;          // 0..7
    const int r0 = blockIdx.y * 32;
    const int c0 = blockIdx.x * 32;

#pragma unroll
    for (int i = 0; i < 4; ++i) {
        int r = ty + i * 8;
        tile[r][tx] = to_bf16_elem(in[(size_t)(r0 + r) * C + (c0 + tx)]);
    }
    __syncthreads();
#pragma unroll
    for (int i = 0; i < 4; ++i) {
        int r = ty + i * 8;
        out[(size_t)(c0 + r) * R + (r0 + tx)] = tile[tx][r];
    }
}

// ---------------------------------------------------------------------------
// Tiled WMMA GEMM:  C[M,N] = A[M,K](bf16) @ Bt^T (+bias) (relu?)
//   Bt is ALWAYS [N,K] row-major (ldb = K-stride): every K-major operand is
//   pre-transposed once, so all staging is coalesced 16B async global->LDS.
// Workgroup: 256 threads (8 wave32) -> 128x128 C tile, K-step 32.
// Wave (2x4 grid): 64x32 region = 4(M) x 2(N) tiles of 16x16.
// Double-buffered LDS. Steady-state body is branch-free:
//   issue next tile's 4 asyncs -> s_wait_asynccnt 4 -> barrier -> 8 WMMA.
// Final K-step is peeled (drain with s_wait_asynccnt 0, no trailing barrier).
// ---------------------------------------------------------------------------
#define TILE_M 128
#define TILE_N 128
#define TILE_K 32

template <bool RELU, bool OUT_BF16>
__global__ __launch_bounds__(256)
void gemm_bf16_wmma(const ushort* __restrict__ A, const ushort* __restrict__ Bt,
                    const float* __restrict__ bias, void* __restrict__ Cout,
                    int M, int N, int K, int lda, int ldb, int ldc)
{
    __shared__ ushort As[2][TILE_M][TILE_K];   // [buf][m][k]  row = 64B
    __shared__ ushort Bs[2][TILE_N][TILE_K];   // [buf][n][k]

    const int tid  = threadIdx.x;
    const int wave = tid >> 5;
    const int lane = tid & 31;
    const int half = lane >> 4;   // 0: lanes 0-15, 1: lanes 16-31
    const int l16  = lane & 15;

    const int bm = blockIdx.y * TILE_M;
    const int bn = blockIdx.x * TILE_N;

    const int wm = (wave & 1) * 64;    // wave row offset inside tile
    const int wn = (wave >> 1) * 32;   // wave col offset inside tile

    // ---- hoisted per-thread staging addresses (advance by constants) ----
    // Tile: 128 rows x 32 cols = 512 x 16B chunks, 2 per thread.
    const int r0c = tid >> 2,          c0c = (tid & 3) * 8;
    const int r1c = (tid + 256) >> 2,  c1c = ((tid + 256) & 3) * 8;
    const ushort* aptr0 = A  + (size_t)(bm + r0c) * lda + c0c;
    const ushort* aptr1 = A  + (size_t)(bm + r1c) * lda + c1c;
    const ushort* bptr0 = Bt + (size_t)(bn + r0c) * ldb + c0c;
    const ushort* bptr1 = Bt + (size_t)(bn + r1c) * ldb + c1c;
    unsigned asl0[2], asl1[2], bsl0[2], bsl1[2];
#pragma unroll
    for (int b = 0; b < 2; ++b) {
        asl0[b] = lds_addr_of(&As[b][r0c][c0c]);
        asl1[b] = lds_addr_of(&As[b][r1c][c1c]);
        bsl0[b] = lds_addr_of(&Bs[b][r0c][c0c]);
        bsl1[b] = lds_addr_of(&Bs[b][r1c][c1c]);
    }

    v8f acc[4][2];
#pragma unroll
    for (int mt = 0; mt < 4; ++mt)
#pragma unroll
        for (int nt = 0; nt < 2; ++nt)
            acc[mt][nt] = 0.0f;

    // ---- WMMA burst over buffer p (ISA 7.12.2 fragment layouts) ----
    auto compute = [&](int p) {
        // A 16x32 bf16: lane(0-15) row=l16, halfs {K0..7, K16..23};
        //               lane(16-31) row=l16, halfs {K8..15, K24..31}
        v16bf afr[4];
#pragma unroll
        for (int mt = 0; mt < 4; ++mt) {
            const ushort* q = &As[p][wm + mt * 16 + l16][half * 8];
            FragU fu;
            fu.u[0] = *(const u32x4*)(q);
            fu.u[1] = *(const u32x4*)(q + 16);
            afr[mt] = fu.v;
        }
        // B 32x16 bf16: lane(0-15) col=l16, halfs {K0..7, K16..23}; lane(16-31) {K8..15, K24..31}
        v16bf bfr[2];
#pragma unroll
        for (int nt = 0; nt < 2; ++nt) {
            const ushort* q = &Bs[p][wn + nt * 16 + l16][half * 8];
            FragU fu;
            fu.u[0] = *(const u32x4*)(q);
            fu.u[1] = *(const u32x4*)(q + 16);
            bfr[nt] = fu.v;
        }
#pragma unroll
        for (int mt = 0; mt < 4; ++mt)
#pragma unroll
            for (int nt = 0; nt < 2; ++nt)
                acc[mt][nt] = __builtin_amdgcn_wmma_f32_16x16x32_bf16(
                    false, afr[mt], false, bfr[nt],
                    (short)0, acc[mt][nt], false, false);
    };

    // ---- prologue: tile 0 -> buffer 0 ----
    async_copy_b128(asl0[0], aptr0); aptr0 += TILE_K;
    async_copy_b128(asl1[0], aptr1); aptr1 += TILE_K;
    async_copy_b128(bsl0[0], bptr0); bptr0 += TILE_K;
    async_copy_b128(bsl1[0], bptr1); bptr1 += TILE_K;

    // ---- steady state: branch-free pipelined loop ----
    const int ksteps = K / TILE_K;
    int p = 0;
    for (int step = 0; step < ksteps - 1; ++step) {
        const int w = p ^ 1;   // prefetch next tile into other buffer
        async_copy_b128(asl0[w], aptr0); aptr0 += TILE_K;
        async_copy_b128(asl1[w], aptr1); aptr1 += TILE_K;
        async_copy_b128(bsl0[w], bptr0); bptr0 += TILE_K;
        async_copy_b128(bsl1[w], bptr1); bptr1 += TILE_K;
        wait_async_le4();      // current tile's 4 retired (in-order)
        __syncthreads();
        compute(p);
        __syncthreads();
        p ^= 1;
    }

    // ---- peeled final K-step: drain and compute (no trailing barrier) ----
    wait_async_le0();
    __syncthreads();
    compute(p);

    // ---- epilogue: f32 C/D layout -> row = half*8 + j, col = l16 ----
#pragma unroll
    for (int mt = 0; mt < 4; ++mt) {
#pragma unroll
        for (int nt = 0; nt < 2; ++nt) {
            const int col = bn + wn + nt * 16 + l16;
            const float bv = bias ? bias[col] : 0.0f;
#pragma unroll
            for (int j = 0; j < 8; ++j) {
                const int row = bm + wm + mt * 16 + half * 8 + j;
                float v = acc[mt][nt][j] + bv;
                if (RELU) v = v > 0.0f ? v : 0.0f;
                if (OUT_BF16)
                    ((ushort*)Cout)[(size_t)row * ldc + col] = f32_to_bf16(v);
                else
                    ((float*)Cout)[(size_t)row * ldc + col] = v;
            }
        }
    }
}

// ---------------------------------------------------------------------------
// Column softmax: h[:,c] = softmax over axis 0 of scores[:,c]*mask[:,c]*scale
// One thread per column -> every row sweep is a coalesced 1KB read.
// ---------------------------------------------------------------------------
__global__ __launch_bounds__(256)
void softmax_col_kernel(const float* __restrict__ scores,
                        const float* __restrict__ mask,
                        ushort* __restrict__ h, int S, float scale)
{
    const int c = blockIdx.x * blockDim.x + threadIdx.x;
    if (c >= S) return;

    float m = -3.0e38f;
    for (int r = 0; r < S; ++r) {
        float l = scores[(size_t)r * S + c] * mask[(size_t)r * S + c] * scale;
        m = fmaxf(m, l);
    }
    float sum = 0.0f;
    for (int r = 0; r < S; ++r) {
        float l = scores[(size_t)r * S + c] * mask[(size_t)r * S + c] * scale;
        float e = __expf(l - m);
        sum += e;
        h[(size_t)r * S + c] = f32_to_bf16(e);
    }
    const float inv = 1.0f / sum;
    for (int r = 0; r < S; ++r) {
        size_t idx = (size_t)r * S + c;
        h[idx] = f32_to_bf16(bf16_to_f32(h[idx]) * inv);
    }
}

// ---------------------------------------------------------------------------
// Fused residual-add + LayerNorm over last dim (D=1024, one row per block).
// Emits f32 result and (optionally) a bf16 copy for the next GEMM.
// ---------------------------------------------------------------------------
__global__ __launch_bounds__(256)
void add_layernorm_kernel(const float* __restrict__ x, const float* __restrict__ res,
                          const float* __restrict__ g, const float* __restrict__ b,
                          float* __restrict__ y, ushort* __restrict__ ybf, int D)
{
    const int row = blockIdx.x;
    const int tid = threadIdx.x;

    __shared__ float red[256];
    __shared__ float s_mu, s_rstd;

    float v[4];
    float s = 0.0f;
#pragma unroll
    for (int i = 0; i < 4; ++i) {
        int c = tid + i * 256;
        float t = x[(size_t)row * D + c] + res[(size_t)row * D + c];
        v[i] = t;
        s += t;
    }
    red[tid] = s;
    __syncthreads();
    for (int off = 128; off > 0; off >>= 1) {
        if (tid < off) red[tid] += red[tid + off];
        __syncthreads();
    }
    if (tid == 0) s_mu = red[0] / (float)D;
    __syncthreads();
    const float mu = s_mu;

    float vs = 0.0f;
#pragma unroll
    for (int i = 0; i < 4; ++i) {
        float d = v[i] - mu;
        vs += d * d;
    }
    red[tid] = vs;
    __syncthreads();
    for (int off = 128; off > 0; off >>= 1) {
        if (tid < off) red[tid] += red[tid + off];
        __syncthreads();
    }
    if (tid == 0) s_rstd = rsqrtf(red[0] / (float)D + 1e-5f);
    __syncthreads();
    const float rstd = s_rstd;

#pragma unroll
    for (int i = 0; i < 4; ++i) {
        int c = tid + i * 256;
        float o = (v[i] - mu) * rstd * g[c] + b[c];
        y[(size_t)row * D + c] = o;
        if (ybf) ybf[(size_t)row * D + c] = f32_to_bf16(o);
    }
}

// ---------------------------------------------------------------------------
// Host-side orchestration
// ---------------------------------------------------------------------------
extern "C" void kernel_launch(void* const* d_in, const int* in_sizes, int n_in,
                              void* d_out, int out_size, void* d_ws, size_t ws_size,
                              hipStream_t stream)
{
    (void)in_sizes; (void)n_in; (void)out_size; (void)ws_size;

    const int S = S_DIM, D = D_DIM, F = F_DIM;

    const float* query = (const float*)d_in[0];
    const float* key   = (const float*)d_in[1];
    const float* value = (const float*)d_in[2];
    const float* mask  = (const float*)d_in[3];
    const float* Wq = (const float*)d_in[4];   const float* bq  = (const float*)d_in[5];
    const float* Wk = (const float*)d_in[6];   const float* bk  = (const float*)d_in[7];
    const float* Wv = (const float*)d_in[8];   const float* bv  = (const float*)d_in[9];
    const float* Wo = (const float*)d_in[10];  const float* bo  = (const float*)d_in[11];
    const float* g1 = (const float*)d_in[12];  const float* be1 = (const float*)d_in[13];
    const float* W1 = (const float*)d_in[14];  const float* bf1 = (const float*)d_in[15];
    const float* W2 = (const float*)d_in[16];  const float* bf2 = (const float*)d_in[17];
    const float* g2 = (const float*)d_in[18];  const float* be2 = (const float*)d_in[19];

    // Workspace bump allocator (deterministic layout every call).
    char*  ws  = (char*)d_ws;
    size_t off = 0;
    auto alloc = [&](size_t bytes) -> void* {
        void* p = ws + off;
        off += (bytes + 255) & ~(size_t)255;
        return p;
    };

    const size_t SD = (size_t)S * D, SS = (size_t)S * S, SF = (size_t)S * F;
    const size_t DD = (size_t)D * D, DF = (size_t)D * F;

    ushort* qin_bf = (ushort*)alloc(SD * 2);
    ushort* kin_bf = (ushort*)alloc(SD * 2);
    ushort* vin_bf = (ushort*)alloc(SD * 2);
    ushort* WqT = (ushort*)alloc(DD * 2);   // [D,D] (out,in) bf16
    ushort* WkT = (ushort*)alloc(DD * 2);
    ushort* WvT = (ushort*)alloc(DD * 2);
    ushort* WoT = (ushort*)alloc(DD * 2);
    ushort* W1T = (ushort*)alloc(DF * 2);   // [F,D]
    ushort* W2T = (ushort*)alloc(DF * 2);   // [D,F]
    ushort* q_bf   = (ushort*)alloc(SD * 2);
    ushort* k_bf   = (ushort*)alloc(SD * 2);
    ushort* v_bf   = (ushort*)alloc(SD * 2);
    ushort* vT     = (ushort*)alloc(SD * 2);   // [D,S]
    float*  scores = (float*) alloc(SS * 4);
    ushort* h_bf   = (ushort*)alloc(SS * 2);
    ushort* hv_bf  = (ushort*)alloc(SD * 2);
    float*  attn   = (float*) alloc(SD * 4);
    float*  n1_f32 = (float*) alloc(SD * 4);
    ushort* n1_bf  = (ushort*)alloc(SD * 2);
    ushort* ff1_bf = (ushort*)alloc(SF * 2);
    float*  ff2    = (float*) alloc(SD * 4);

    const dim3 blk(256);

    // --- activations: convert to bf16 (row-major, A-side) ---
    cvt_f32_to_bf16<<<512, blk, 0, stream>>>(query, qin_bf, (int)SD);
    cvt_f32_to_bf16<<<512, blk, 0, stream>>>(key,   kin_bf, (int)SD);
    cvt_f32_to_bf16<<<512, blk, 0, stream>>>(value, vin_bf, (int)SD);

    // --- weights: fused convert + transpose to [N,K] bf16 (B-side) ---
    transpose_to_bf16<float><<<dim3(D / 32, D / 32), blk, 0, stream>>>(Wq, WqT, D, D);
    transpose_to_bf16<float><<<dim3(D / 32, D / 32), blk, 0, stream>>>(Wk, WkT, D, D);
    transpose_to_bf16<float><<<dim3(D / 32, D / 32), blk, 0, stream>>>(Wv, WvT, D, D);
    transpose_to_bf16<float><<<dim3(D / 32, D / 32), blk, 0, stream>>>(Wo, WoT, D, D);
    transpose_to_bf16<float><<<dim3(F / 32, D / 32), blk, 0, stream>>>(W1, W1T, D, F);
    transpose_to_bf16<float><<<dim3(D / 32, F / 32), blk, 0, stream>>>(W2, W2T, F, D);

    const dim3 gSD(D / TILE_N, S / TILE_M);   // [S,D] outputs
    const dim3 gSS(S / TILE_N, S / TILE_M);   // [S,S] outputs
    const dim3 gSF(F / TILE_N, S / TILE_M);   // [S,F] outputs

    // --- QKV projections: bf16 out for downstream WMMA ---
    gemm_bf16_wmma<false, true><<<gSD, blk, 0, stream>>>(
        qin_bf, WqT, bq, q_bf, S, D, D, D, D, D);
    gemm_bf16_wmma<false, true><<<gSD, blk, 0, stream>>>(
        kin_bf, WkT, bk, k_bf, S, D, D, D, D, D);
    gemm_bf16_wmma<false, true><<<gSD, blk, 0, stream>>>(
        vin_bf, WvT, bv, v_bf, S, D, D, D, D, D);

    // --- scores = q @ k^T  (k_bf is already [N,K]; f32 out for softmax) ---
    gemm_bf16_wmma<false, false><<<gSS, blk, 0, stream>>>(
        q_bf, k_bf, nullptr, scores, S, S, D, D, D, S);

    // --- h = softmax(scores * mask / sqrt(D), axis=0), bf16 out ---
    softmax_col_kernel<<<S / 256, blk, 0, stream>>>(
        scores, mask, h_bf, S, 1.0f / sqrtf((float)D));

    // --- v^T for the h @ v GEMM ---
    transpose_to_bf16<ushort><<<dim3(D / 32, S / 32), blk, 0, stream>>>(v_bf, vT, S, D);

    // --- attn = (h @ v) @ Wo + bo ---
    gemm_bf16_wmma<false, true><<<gSD, blk, 0, stream>>>(
        h_bf, vT, nullptr, hv_bf, S, D, S, S, S, D);
    gemm_bf16_wmma<false, false><<<gSD, blk, 0, stream>>>(
        hv_bf, WoT, bo, attn, S, D, D, D, D, D);

    // --- n1 = LayerNorm(attn + query) ---
    add_layernorm_kernel<<<S, blk, 0, stream>>>(
        attn, query, g1, be1, n1_f32, n1_bf, D);

    // --- ff = relu(n1 @ W1 + bf1) @ W2 + bf2 ---
    gemm_bf16_wmma<true, true><<<gSF, blk, 0, stream>>>(
        n1_bf, W1T, bf1, ff1_bf, S, F, D, D, D, F);
    gemm_bf16_wmma<false, false><<<gSD, blk, 0, stream>>>(
        ff1_bf, W2T, bf2, ff2, S, D, F, F, F, D);

    // --- out = LayerNorm(ff + n1) ---
    add_layernorm_kernel<<<S, blk, 0, stream>>>(
        ff2, n1_f32, g2, be2, (float*)d_out, nullptr, D);
}
